// GAT_6622839571297
// MI455X (gfx1250) — matrix operand order repaired and checked by reading, not compiled, not run on hardware
//
#include <hip/hip_runtime.h>

typedef __attribute__((ext_vector_type(16))) __bf16          v16bf;
typedef __attribute__((ext_vector_type(16))) unsigned short  v16us;
typedef __attribute__((ext_vector_type(8)))  unsigned short  v8us;
typedef __attribute__((ext_vector_type(8)))  float           v8f;

#define NEG_SLOPE 0.2f
#define EPSV      1e-16f

__device__ __forceinline__ unsigned short f2bf(float f) {
  unsigned u = __float_as_uint(f);
  u += 0x7FFFu + ((u >> 16) & 1u);          // round-to-nearest-even
  return (unsigned short)(u >> 16);
}
// monotonic float -> uint key (total order incl. negatives)
__device__ __forceinline__ unsigned fkey(float f) {
  unsigned u = __float_as_uint(f);
  return (u & 0x80000000u) ? ~u : (u | 0x80000000u);
}
__device__ __forceinline__ float fdecode(unsigned k) {
  return (k & 0x80000000u) ? __uint_as_float(k & 0x7FFFFFFFu)
                           : __uint_as_float(~k);
}
__device__ __forceinline__ void edge_ends(const int* __restrict__ ei,
                                          long long E, long long e,
                                          int& s, int& d) {
  if (e < E) { s = ei[e]; d = ei[E + e]; }       // edge_index rows [src; dst]
  else       { s = d = (int)(e - E); }           // appended self loops
}

// ---------------------------------------------------------------- utilities
__global__ void k_fill0(uint4* __restrict__ p, long long n4) {
  long long i = (long long)blockIdx.x * blockDim.x + threadIdx.x;
  if (i < n4) p[i] = uint4{0u, 0u, 0u, 0u};
}
__global__ void k_cvt_bf16(const float* __restrict__ x,
                           unsigned short* __restrict__ y, long long n) {
  long long i = (long long)blockIdx.x * blockDim.x + threadIdx.x;
  if (i < n) y[i] = f2bf(x[i]);
}
// W:[K][Ncol] row-major  ->  Wt:[Ncol][K] bf16 (so B columns are contiguous)
__global__ void k_transpose_bf16(const float* __restrict__ W,
                                 unsigned short* __restrict__ Wt,
                                 int K, int Ncol) {
  long long i = (long long)blockIdx.x * blockDim.x + threadIdx.x;
  if (i < (long long)K * Ncol) {
    int k = (int)(i % K);
    int n = (int)(i / K);
    Wt[(long long)n * K + k] = f2bf(W[(long long)k * Ncol + n]);
  }
}

// ------------------------------------------------------------ WMMA bf16 GEMM
// C[M][Ncol] = A[M][K] (bf16) * Bt[Ncol][K]^T (bf16), f32 accumulate.
// One wave computes a 16x64 output strip: 4 accumulators share one A fragment
// per 32-wide K step (4x less A traffic, 2.5 loads/WMMA instead of 4).
__device__ __forceinline__ v16bf load_a_frag(const unsigned short* __restrict__ p) {
  v8us a0 = *(const v8us*)(p);         // K run 0..7  (+half*16)
  v8us a1 = *(const v8us*)(p + 8);     // K run 8..15 (+half*16)
  v16us av = __builtin_shufflevector(a0, a1, 0, 1, 2, 3, 4, 5, 6, 7,
                                             8, 9, 10, 11, 12, 13, 14, 15);
  return __builtin_bit_cast(v16bf, av);
}
__device__ __forceinline__ v16bf load_b_frag(const unsigned short* __restrict__ p) {
  v16us bv = *(const v16us*)(p);       // 16 contiguous K for this column
  return __builtin_bit_cast(v16bf, bv);
}

__global__ void __launch_bounds__(256)
k_gemm_bf16_wmma(const unsigned short* __restrict__ A,
                 const unsigned short* __restrict__ Bt,
                 float* __restrict__ C, int M, int K, int Ncol) {
  int wave  = blockIdx.x * (blockDim.x >> 5) + (threadIdx.x >> 5);
  int lane  = threadIdx.x & 31;
  int strips = Ncol >> 6;                       // 64-wide output strips
  int totalWaves = (M >> 4) * strips;
  if (wave >= totalWaves) return;
  int tm  = (wave / strips) << 4;
  int tn0 = (wave % strips) << 6;

  int half = lane >> 4;   // lane half selects K 16*half..+15 (contiguous, same for A and B)
  int l15  = lane & 15;
  const unsigned short* arow = A + (size_t)(tm + l15) * K + (half ? 16 : 0);
  const unsigned short* bc0 = Bt + (size_t)(tn0 + l15)      * K + (half ? 16 : 0);
  const unsigned short* bc1 = Bt + (size_t)(tn0 + 16 + l15) * K + (half ? 16 : 0);
  const unsigned short* bc2 = Bt + (size_t)(tn0 + 32 + l15) * K + (half ? 16 : 0);
  const unsigned short* bc3 = Bt + (size_t)(tn0 + 48 + l15) * K + (half ? 16 : 0);

  v8f acc0 = {}, acc1 = {}, acc2 = {}, acc3 = {};
  for (int kb = 0; kb < K; kb += 32) {
    v16bf a  = load_a_frag(arow + kb);
    v16bf b0 = load_b_frag(bc0 + kb);
    v16bf b1 = load_b_frag(bc1 + kb);
    v16bf b2 = load_b_frag(bc2 + kb);
    v16bf b3 = load_b_frag(bc3 + kb);
    acc0 = __builtin_amdgcn_wmma_f32_16x16x32_bf16(false, a, false, b0,
                                                   (short)0, acc0, false, false);
    acc1 = __builtin_amdgcn_wmma_f32_16x16x32_bf16(false, a, false, b1,
                                                   (short)0, acc1, false, false);
    acc2 = __builtin_amdgcn_wmma_f32_16x16x32_bf16(false, a, false, b2,
                                                   (short)0, acc2, false, false);
    acc3 = __builtin_amdgcn_wmma_f32_16x16x32_bf16(false, a, false, b3,
                                                   (short)0, acc3, false, false);
  }
  int mb = tm + (half ? 8 : 0);                 // C/D layout: VGPR e -> M=e(+8)
  float* crow = C + (size_t)mb * Ncol + tn0 + l15;
#pragma unroll
  for (int e = 0; e < 8; ++e) {
    float* r = crow + (size_t)e * Ncol;
    r[0] = acc0[e]; r[16] = acc1[e]; r[32] = acc2[e]; r[48] = acc3[e];
  }
}

// -------------------------------------------------- attention projections
__global__ void k_att_dot(const float* __restrict__ Hf,
                          const float* __restrict__ att_s,
                          const float* __restrict__ att_d,
                          float* __restrict__ as, float* __restrict__ ad,
                          long long N, int H, int Cc) {
  long long i = (long long)blockIdx.x * blockDim.x + threadIdx.x;
  if (i >= N * (long long)H) return;
  int h = (int)(i % H);
  long long n = i / H;
  const float* hp = Hf + n * (long long)(H * Cc) + (long long)h * Cc;
  const float* sp = att_s + (long long)h * Cc;
  const float* dp = att_d + (long long)h * Cc;
  float s = 0.f, d = 0.f;
  for (int c = 0; c < Cc; ++c) { float v = hp[c]; s += v * sp[c]; d += v * dp[c]; }
  as[i] = s; ad[i] = d;
}

// -------------------------------------------------- edge softmax passes
__global__ void k_edge_max(const int* __restrict__ ei, long long E, long long Etot,
                           const float* __restrict__ as, const float* __restrict__ ad,
                           unsigned* __restrict__ mkey, int H) {
  long long i = (long long)blockIdx.x * blockDim.x + threadIdx.x;
  if (i >= Etot * (long long)H) return;
  int h = (int)(i % H);
  long long e = i / H;
  int s, d; edge_ends(ei, E, e, s, d);
  float v = as[(long long)s * H + h] + ad[(long long)d * H + h];
  v = v > 0.f ? v : NEG_SLOPE * v;
  atomicMax(&mkey[(long long)d * H + h], fkey(v));
}

__global__ void k_edge_sum(const int* __restrict__ ei, long long E, long long Etot,
                           const float* __restrict__ as, const float* __restrict__ ad,
                           const unsigned* __restrict__ mkey,
                           float* __restrict__ ssum, int H) {
  long long i = (long long)blockIdx.x * blockDim.x + threadIdx.x;
  if (i >= Etot * (long long)H) return;
  int h = (int)(i % H);
  long long e = i / H;
  int s, d; edge_ends(ei, E, e, s, d);
  long long dH = (long long)d * H + h;
  float v = as[(long long)s * H + h] + ad[dH];
  v = v > 0.f ? v : NEG_SLOPE * v;
  float ex = __expf(v - fdecode(mkey[dH]));
  atomicAdd(&ssum[dH], ex);
}

// one block per edge; thread t = h*Cc + c covers all H*Cc feature channels
__global__ void k_edge_agg(const int* __restrict__ ei, long long E,
                           const float* __restrict__ as, const float* __restrict__ ad,
                           const unsigned* __restrict__ mkey,
                           const float* __restrict__ ssum,
                           const float* __restrict__ Hf, float* __restrict__ agg,
                           int H, int Cc) {
  long long e = blockIdx.x;
  int t = threadIdx.x;
  int h = t / Cc;
  int s, d; edge_ends(ei, E, e, s, d);
  long long dH = (long long)d * H + h;
  float v = as[(long long)s * H + h] + ad[dH];
  v = v > 0.f ? v : NEG_SLOPE * v;
  float alpha = __expf(v - fdecode(mkey[dH])) / (ssum[dH] + EPSV);
  int F = H * Cc;
  atomicAdd(&agg[(long long)d * F + t], alpha * Hf[(long long)s * F + t]);
}

// bias + ELU + bf16 convert (layer-1 output feeds GEMM2)
__global__ void k_epilogue1(const float* __restrict__ agg,
                            const float* __restrict__ b,
                            unsigned short* __restrict__ actb,
                            long long n, int F) {
  long long i = (long long)blockIdx.x * blockDim.x + threadIdx.x;
  if (i >= n) return;
  float v = agg[i] + b[i % F];
  v = v > 0.f ? v : (__expf(v) - 1.f);          // ELU(alpha=1)
  actb[i] = f2bf(v);
}

__global__ void k_bias_out(float* __restrict__ out, const float* __restrict__ b,
                           long long n, int F) {
  long long i = (long long)blockIdx.x * blockDim.x + threadIdx.x;
  if (i < n) out[i] += b[i % F];
}

static inline unsigned gsz(long long n, int b) { return (unsigned)((n + b - 1) / b); }

extern "C" void kernel_launch(void* const* d_in, const int* in_sizes, int n_in,
                              void* d_out, int out_size, void* d_ws, size_t ws_size,
                              hipStream_t stream) {
  const float* x   = (const float*)d_in[0];
  const int*   ei  = (const int*)d_in[1];
  const float* W1  = (const float*)d_in[2];
  const float* as1 = (const float*)d_in[3];
  const float* ad1 = (const float*)d_in[4];
  const float* b1  = (const float*)d_in[5];
  const float* W2  = (const float*)d_in[6];
  const float* as2 = (const float*)d_in[7];
  const float* ad2 = (const float*)d_in[8];
  const float* b2  = (const float*)d_in[9];
  float* out = (float*)d_out;

  const long long N = 50000, E = 800000, Etot = E + N;
  const int IN = 128, F1 = 256, H1 = 8, C1 = 32, F2 = 64;

  char* ws = (char*)d_ws;
  auto carve = [&](size_t bytes) -> char* {
    char* p = ws; ws += (bytes + 255) & ~(size_t)255; return p;
  };
  unsigned short* xb    = (unsigned short*)carve((size_t)N * IN * 2);
  unsigned short* W1t   = (unsigned short*)carve((size_t)F1 * IN * 2);
  float*          Hf1   = (float*)carve((size_t)N * F1 * 4);
  float*          aS1   = (float*)carve((size_t)N * H1 * 4);
  float*          aD1   = (float*)carve((size_t)N * H1 * 4);
  unsigned*       mk1   = (unsigned*)carve((size_t)N * H1 * 4);
  float*          sm1   = (float*)carve((size_t)N * H1 * 4);
  float*          agg1  = (float*)carve((size_t)N * F1 * 4);
  unsigned short* act1b = (unsigned short*)carve((size_t)N * F1 * 2);
  unsigned short* W2t   = (unsigned short*)carve((size_t)F2 * F1 * 2);
  float*          Hf2   = (float*)carve((size_t)N * F2 * 4);
  float*          aS2   = (float*)carve((size_t)N * 4);
  float*          aD2   = (float*)carve((size_t)N * 4);
  unsigned*       mk2   = (unsigned*)carve((size_t)N * 4);
  float*          sm2   = (float*)carve((size_t)N * 4);

  // ---- zero init (keys: 0 < fkey(x) for all finite x; sums/agg/out: 0.0f)
  k_fill0<<<gsz(N * H1 / 4, 256), 256, 0, stream>>>((uint4*)mk1,  N * H1 / 4);
  k_fill0<<<gsz(N * H1 / 4, 256), 256, 0, stream>>>((uint4*)sm1,  N * H1 / 4);
  k_fill0<<<gsz(N * F1 / 4, 256), 256, 0, stream>>>((uint4*)agg1, N * F1 / 4);
  k_fill0<<<gsz(N / 4, 256), 256, 0, stream>>>((uint4*)mk2, N / 4);
  k_fill0<<<gsz(N / 4, 256), 256, 0, stream>>>((uint4*)sm2, N / 4);
  k_fill0<<<gsz(N * F2 / 4, 256), 256, 0, stream>>>((uint4*)out, N * F2 / 4);

  // ---- layer 1: GEMM (WMMA bf16), projections, edge softmax, aggregate
  k_cvt_bf16<<<gsz(N * IN, 256), 256, 0, stream>>>(x, xb, N * IN);
  k_transpose_bf16<<<gsz((long long)IN * F1, 256), 256, 0, stream>>>(W1, W1t, IN, F1);
  k_gemm_bf16_wmma<<<gsz((N / 16) * (F1 / 64), 8), 256, 0, stream>>>(
      xb, W1t, Hf1, (int)N, IN, F1);
  k_att_dot<<<gsz(N * H1, 256), 256, 0, stream>>>(Hf1, as1, ad1, aS1, aD1, N, H1, C1);
  k_edge_max<<<gsz(Etot * H1, 256), 256, 0, stream>>>(ei, E, Etot, aS1, aD1, mk1, H1);
  k_edge_sum<<<gsz(Etot * H1, 256), 256, 0, stream>>>(ei, E, Etot, aS1, aD1, mk1, sm1, H1);
  k_edge_agg<<<(unsigned)Etot, H1 * C1, 0, stream>>>(ei, E, aS1, aD1, mk1, sm1,
                                                     Hf1, agg1, H1, C1);
  k_epilogue1<<<gsz(N * F1, 256), 256, 0, stream>>>(agg1, b1, act1b, N * F1, F1);

  // ---- layer 2
  k_transpose_bf16<<<gsz((long long)F1 * F2, 256), 256, 0, stream>>>(W2, W2t, F1, F2);
  k_gemm_bf16_wmma<<<gsz((N / 16) * (F2 / 64), 8), 256, 0, stream>>>(
      act1b, W2t, Hf2, (int)N, F1, F2);
  k_att_dot<<<gsz(N, 256), 256, 0, stream>>>(Hf2, as2, ad2, aS2, aD2, N, 1, F2);
  k_edge_max<<<gsz(Etot, 256), 256, 0, stream>>>(ei, E, Etot, aS2, aD2, mk2, 1);
  k_edge_sum<<<gsz(Etot, 256), 256, 0, stream>>>(ei, E, Etot, aS2, aD2, mk2, sm2, 1);
  k_edge_agg<<<(unsigned)Etot, F2, 0, stream>>>(ei, E, aS2, aD2, mk2, sm2,
                                                Hf2, out, 1, F2);
  k_bias_out<<<gsz(N * F2, 256), 256, 0, stream>>>(out, b2, N * F2, F2);
}
